// ContrastiveLoss_79534204387857
// MI455X (gfx1250) — matrix operand order repaired and checked by reading, not compiled, not run on hardware
//
#include <hip/hip_runtime.h>
#include <hip/hip_bf16.h>

typedef __attribute__((ext_vector_type(16))) _Float16 v16h;
typedef __attribute__((ext_vector_type(8)))  _Float16 v8h;
typedef __attribute__((ext_vector_type(4)))  _Float16 h4;
typedef __attribute__((ext_vector_type(8)))  float    v8f;

namespace {
constexpr int kBatch = 4096;
constexpr int kDim   = 128;
constexpr int kN2    = 8192;            // 2 * kBatch
constexpr int kColChunks = 8;           // column-dimension grid split
constexpr int kColsPerChunk = kN2 / kColChunks;  // 1024 columns per chunk
}

union AFrag16 {
    v16h v;
    v8h  h[2];
};

// ---------------------------------------------------------------------------
// Kernel 1: L2-normalize each of the 8192 rows (emb_i rows 0..4095, emb_j
// rows 4096..8191). One wave32 per row; lane holds 4 floats. Writes both an
// f32 copy (for exact positive dots) and an f16 copy (for WMMA Gram).
// ---------------------------------------------------------------------------
__global__ void nrm_kernel(const float* __restrict__ ei,
                           const float* __restrict__ ej,
                           float* __restrict__ z32,
                           _Float16* __restrict__ z16) {
    const int lane = threadIdx.x & 31;
    const int wave = threadIdx.x >> 5;
    const int r    = blockIdx.x * 8 + wave;              // 8 waves per block

    const float* src = (r < kBatch) ? (ei + r * kDim)
                                    : (ej + (r - kBatch) * kDim);
    float4 v = ((const float4*)src)[lane];
    float ss = v.x * v.x + v.y * v.y + v.z * v.z + v.w * v.w;
#pragma unroll
    for (int m = 16; m >= 1; m >>= 1) ss += __shfl_xor(ss, m, 32);

    const float scale = 1.0f / fmaxf(sqrtf(ss), 1e-12f);
    v.x *= scale; v.y *= scale; v.z *= scale; v.w *= scale;

    ((float4*)(z32 + r * kDim))[lane] = v;
    h4 h;
    h.x = (_Float16)v.x; h.y = (_Float16)v.y;
    h.z = (_Float16)v.z; h.w = (_Float16)v.w;
    ((h4*)(z16 + r * kDim))[lane] = h;
}

// ---------------------------------------------------------------------------
// Kernel 2: positive-pair dots pos[k] = z_i[k] . z_j[k] (f32 precision).
// One wave32 per pair.
// ---------------------------------------------------------------------------
__global__ void pos_kernel(const float* __restrict__ z32,
                           float* __restrict__ posdot) {
    const int lane = threadIdx.x & 31;
    const int wave = threadIdx.x >> 5;
    const int k    = blockIdx.x * 8 + wave;

    float4 a = ((const float4*)(z32 + k * kDim))[lane];
    float4 b = ((const float4*)(z32 + (k + kBatch) * kDim))[lane];
    float d = a.x * b.x + a.y * b.y + a.z * b.z + a.w * b.w;
#pragma unroll
    for (int m = 16; m >= 1; m >>= 1) d += __shfl_xor(d, m, 32);
    if (lane == 0) posdot[k] = d;
}

// ---------------------------------------------------------------------------
// Kernel 3: fused Gram + exp + row-reduce. Block = 128 threads = 4 waves,
// each wave owns a distinct 16-row tile (A resident in VGPRs for the whole
// sweep). Grid = (8192/64 row blocks, 8 column chunks). Each wave sweeps
// 1024 columns in 16-wide tiles: 4 chained v_wmma_f32_16x16x32_f16 over
// K=128, then exp(2*sim) accumulated into per-row partial sums written to
// rowsum[chunk][row] (deterministic 2-D partials; no float atomics).
// ---------------------------------------------------------------------------
__global__ void gram_kernel(const _Float16* __restrict__ z16,
                            float* __restrict__ rowsum) {
    const int lane = threadIdx.x & 31;
    const int wave = threadIdx.x >> 5;
    const int l15  = lane & 15;
    const int half = lane >> 4;                 // 0 for lanes 0-15, 1 for 16-31
    const int ra   = blockIdx.x * 64 + wave * 16;  // this wave's 16-row tile

    // --- A fragments: 16x32 f16 per K-chunk, ISA 7.12.2 layout.
    // lane l: row ra+(l&15); halfs [kb+8h .. +7] then [kb+16+8h .. +7]
    const _Float16* zr = z16 + (ra + l15) * kDim;
    AFrag16 a[4];
#pragma unroll
    for (int kk = 0; kk < 4; ++kk) {
        const int kb = kk * 32;
        a[kk].h[0] = *(const v8h*)(zr + kb + 8 * half);
        a[kk].h[1] = *(const v8h*)(zr + kb + 16 + 8 * half);
    }

    const int cb0 = blockIdx.y * kColsPerChunk;
    float rs[8] = {0.f, 0.f, 0.f, 0.f, 0.f, 0.f, 0.f, 0.f};

    for (int cb = cb0; cb < cb0 + kColsPerChunk; cb += 16) {
        // --- B fragment source: column n of B = row cb+n of z16.
        const _Float16* zc = z16 + (cb + l15) * kDim;
        if (cb + 16 < cb0 + kColsPerChunk) {
            // hint next tile into cache (global_prefetch_b8)
            __builtin_prefetch(z16 + (cb + 16 + l15) * kDim, 0, 0);
        }

        v8f c = {};
#pragma unroll
        for (int kk = 0; kk < 4; ++kk) {
            const int kb = kk * 32;
            // B 32x16 f16 layout: lane l holds K = 8*half..8*half+7,
            // then 16+8*half..16+8*half+7 (same order as A).
            AFrag16 b;
            b.h[0] = *(const v8h*)(zc + kb + 8 * half);
            b.h[1] = *(const v8h*)(zc + kb + 16 + 8 * half);
            c = __builtin_amdgcn_wmma_f32_16x16x32_f16(
                    /*neg_a=*/false, a[kk].v,
                    /*neg_b=*/false, b.v,
                    /*c_mod=*/(short)0, c,
                    /*reuse_a=*/false, /*reuse_b=*/false);
        }
        // exp(sim / 0.5) accumulated per (row, lane-column) slot.
        // C/D layout: VGPR g -> row ra + g + 8*half, col cb + (l&15).
#pragma unroll
        for (int g = 0; g < 8; ++g) rs[g] += __expf(2.0f * c[g]);
    }

    // Reduce the 16 column-lanes of each half-wave.
#pragma unroll
    for (int m = 8; m >= 1; m >>= 1) {
#pragma unroll
        for (int g = 0; g < 8; ++g) rs[g] += __shfl_xor(rs[g], m, 32);
    }
    if (l15 == 0) {
        float* dst = rowsum + blockIdx.y * kN2 + ra + 8 * half;
#pragma unroll
        for (int g = 0; g < 8; ++g) dst[g] = rs[g];
    }
}

// ---------------------------------------------------------------------------
// Kernel 4: final reduction.
// denom[r] = sum_chunks rowsum[chunk][r] - e^2   (removes diagonal term)
// loss = ( sum_r log(denom[r]) - 4 * sum_k pos[k] ) / (2*B*5)
// ---------------------------------------------------------------------------
__global__ void loss_kernel(const float* __restrict__ rowsum,
                            const float* __restrict__ posdot,
                            float* __restrict__ out) {
    const int tid = threadIdx.x;
    const float e2 = __expf(2.0f);
    float t = 0.0f;
    for (int r = tid; r < kN2; r += 256) {
        float d = 0.0f;
#pragma unroll
        for (int ch = 0; ch < kColChunks; ++ch) d += rowsum[ch * kN2 + r];
        t += logf(d - e2);
    }
    for (int k = tid; k < kBatch; k += 256) t -= 4.0f * posdot[k];

    __shared__ float red[256];
    red[tid] = t;
    __syncthreads();
    for (int s = 128; s > 0; s >>= 1) {
        if (tid < s) red[tid] += red[tid + s];
        __syncthreads();
    }
    if (tid == 0) out[0] = red[0] / (float)(2 * kBatch * 5);
}

// ---------------------------------------------------------------------------
extern "C" void kernel_launch(void* const* d_in, const int* in_sizes, int n_in,
                              void* d_out, int out_size, void* d_ws, size_t ws_size,
                              hipStream_t stream) {
    (void)in_sizes; (void)n_in; (void)out_size; (void)ws_size;
    const float* emb_i = (const float*)d_in[0];
    const float* emb_j = (const float*)d_in[1];
    float* out = (float*)d_out;

    // Workspace layout (~6.5 MB):
    //   z32    : 8192*128 f32  (4 MB)
    //   z16    : 8192*128 f16  (2 MB)
    //   posdot : 4096 f32
    //   rowsum : 8 * 8192 f32 partial sums
    float*     z32    = (float*)d_ws;
    _Float16*  z16    = (_Float16*)(z32 + (size_t)kN2 * kDim);
    float*     posdot = (float*)(z16 + (size_t)kN2 * kDim);
    float*     rowsum = posdot + kBatch;

    nrm_kernel<<<kN2 / 8, 256, 0, stream>>>(emb_i, emb_j, z32, z16);
    pos_kernel<<<kBatch / 8, 256, 0, stream>>>(z32, posdot);
    gram_kernel<<<dim3(kN2 / 64, kColChunks), 128, 0, stream>>>(z16, rowsum);
    loss_kernel<<<1, 256, 0, stream>>>(rowsum, posdot, out);
}
